// HybridGDN_28527172780880
// MI455X (gfx1250) — matrix operand order repaired and checked by reading, not compiled, not run on hardware
//
#include <hip/hip_runtime.h>

// ---------------------------------------------------------------------------
// Types for CDNA5 WMMA (wave32): v_wmma_f32_16x16x32_bf16
// ---------------------------------------------------------------------------
typedef __attribute__((ext_vector_type(16))) __bf16          bf16x16;
typedef __attribute__((ext_vector_type(8)))  float           f32x8;
typedef __attribute__((ext_vector_type(16))) unsigned short  u16x16;
typedef __attribute__((ext_vector_type(4)))  unsigned int    u32x4;
typedef __attribute__((ext_vector_type(8)))  unsigned int    u32x8;

__device__ __forceinline__ unsigned short f2bf(float f) {
  union { float f; unsigned u; } c; c.f = f;
  unsigned u = c.u;
  return (unsigned short)((u + 0x7FFFu + ((u >> 16) & 1u)) >> 16); // RNE
}

// ---------------------------------------------------------------------------
// bf16 WMMA GEMM: C[M,N] = A[M,K] @ W[K,N], both inputs pre-converted bf16.
//   A  : [M][K] bf16 row-major
//   Wt : [N][K] bf16 (pre-transposed weight)
// Tile 64x64x32, 256 threads = 8 waves, each wave owns two 16x16 C subtiles.
// A tile  -> LDS via per-lane GLOBAL_LOAD_ASYNC_TO_LDS_B128 (ASYNCcnt)
// B tile  -> LDS via Tensor Data Mover TENSOR_LOAD_TO_LDS   (TENSORcnt)
// ---------------------------------------------------------------------------
#define TM 64
#define TN 64
#define TK 32
#define LROW (TK + 8)                    // LDS row pitch in ushorts (80 B)

__global__ __launch_bounds__(256) void hg_gemm(
    const unsigned short* __restrict__ A,
    const unsigned short* __restrict__ Wt,
    float* __restrict__ C, int M, int N, int K)
{
  // single LDS block so byte offsets are known (group segment starts at 0)
  __shared__ unsigned short smem[TM * LROW + TN * LROW];
  const unsigned BS_OFF_B = TM * LROW * 2;      // byte offset of B region

  const int tid  = threadIdx.x;
  const int lane = tid & 31;
  const int wave = tid >> 5;            // 0..7
  const int m0 = blockIdx.y * TM;
  const int n0 = blockIdx.x * TN;

  const int r    = wave & 3;            // subtile row  (0..3)
  const int c0   = (wave >> 2) * 2;     // subtile cols (c0, c0+1)
  const int half = (lane >> 4) & 1;
  const int l15  = lane & 15;

  const int arow = tid >> 2;            // staging row (0..63)
  const int akb  = (tid & 3) * 8;       // staging k-chunk (0,8,16,24)
  const unsigned lda = (unsigned)((arow * LROW + akb) * 2);   // LDS byte addr

  f32x8 acc0 = {}; f32x8 acc1 = {};

  for (int kk = 0; kk < K; kk += TK) {
    // ---- A tile: per-lane async DMA, 16 bytes each (64x32 bf16 total)
    {
      const unsigned short* ga = A + (size_t)(m0 + arow) * K + kk + akb;
      asm volatile("global_load_async_to_lds_b128 %0, %1, off"
                   :: "v"(lda), "v"(ga) : "memory");
    }
    // ---- B tile: TDM, one descriptor from wave 0 (EXEC ignored by TDM)
    if (wave == 0) {
      unsigned long long ga = (unsigned long long)(const void*)
                              (Wt + (size_t)n0 * K + kk);
      u32x4 g0;
      g0[0] = 1u;                                        // count=1, user mode
      g0[1] = BS_OFF_B;                                  // lds_addr
      g0[2] = (unsigned)(ga & 0xffffffffu);              // global_addr[31:0]
      g0[3] = (unsigned)((ga >> 32) & 0x01ffffffu)       // global_addr[56:32]
              | 0x80000000u;                             // type=2 ("image")
      const unsigned td0 = (unsigned)K;                  // tensor dim0 (cols)
      const unsigned td1 = (unsigned)(N - n0);           // rows left -> OOB=0
      const unsigned long long st0 = (unsigned long long)K;
      u32x8 g1;
      g1[0] = (1u << 16)            // data_size = 1 (2 bytes)
            | (1u << 20)            // pad_enable
            | (3u << 22)            // pad_interval: 16 DWORDs (one 64B row)
            | (3u << 25);           // pad_amount:   4 DWORDs (16 B)
      g1[1] = (td0 & 0xffffu) << 16;                     // tensor_dim0[15:0]
      g1[2] = (td0 >> 16) | ((td1 & 0xffffu) << 16);     // dim0 hi | dim1 lo
      g1[3] = (td1 >> 16) | ((unsigned)TK << 16);        // dim1 hi | tile_dim0
      g1[4] = (unsigned)TN;                              // tile_dim1 | tile_dim2=0
      g1[5] = (unsigned)(st0 & 0xffffffffu);             // dim0_stride[31:0]
      g1[6] = (unsigned)((st0 >> 32) & 0xffffu);         // dim0_stride[47:32]
      g1[7] = 0u;
      asm volatile("tensor_load_to_lds %0, %1" :: "s"(g0), "s"(g1) : "memory");
      asm volatile("s_wait_tensorcnt 0x0" ::: "memory");
    }
    if (kk + TK < K)
      __builtin_prefetch(A + (size_t)(m0 + arow) * K + kk + TK + akb, 0, 1);
    asm volatile("s_wait_asynccnt 0x0" ::: "memory");
    __syncthreads();

    // ---- A fragment (16x32): lane<16 -> K 0-15 ; lane>=16 -> K 16-31
    const unsigned short* As = smem;
    const unsigned short* Bs = smem + TM * LROW;
    const int fr = r * 16 + l15;
    const int ak = half * 16;
    u16x16 av;
    #pragma unroll
    for (int i = 0; i < 8; ++i) {
      av[i]     = As[fr * LROW + ak + i];
      av[8 + i] = As[fr * LROW + ak + 8 + i];
    }
    bf16x16 af = __builtin_bit_cast(bf16x16, av);

    // ---- B fragments (32x16): lane<16 -> K 0-15 ; lane>=16 -> K 16-31
    const int bk = half * 16;
    u16x16 bv0, bv1;
    {
      const unsigned short* p0 = &Bs[(c0 * 16 + l15) * LROW + bk];
      const unsigned short* p1 = &Bs[((c0 + 1) * 16 + l15) * LROW + bk];
      #pragma unroll
      for (int i = 0; i < 16; ++i) { bv0[i] = p0[i]; bv1[i] = p1[i]; }
    }
    bf16x16 bf0 = __builtin_bit_cast(bf16x16, bv0);
    bf16x16 bf1 = __builtin_bit_cast(bf16x16, bv1);

    acc0 = __builtin_amdgcn_wmma_f32_16x16x32_bf16(false, af, false, bf0, (short)0, acc0, false, false);
    acc1 = __builtin_amdgcn_wmma_f32_16x16x32_bf16(false, af, false, bf1, (short)0, acc1, false, false);
    __syncthreads();
  }

  // ---- store C (16x16 f32 layout: lane = col, vgpr i = row (+8 for hi half))
  const int crow0 = m0 + r * 16 + half * 8;
  const int ccol0 = n0 + c0 * 16 + l15;
  const int ccol1 = n0 + (c0 + 1) * 16 + l15;
  #pragma unroll
  for (int i = 0; i < 8; ++i) {
    const size_t row = (size_t)(crow0 + i);
    if (ccol0 < N) C[row * N + ccol0] = acc0[i];
    if (ccol1 < N) C[row * N + ccol1] = acc1[i];
  }
}

// ---------------------------------------------------------------------------
// Weight convert + transpose: Wt[n][k] = bf16(W[k][n])  (once per launch)
// ---------------------------------------------------------------------------
__global__ __launch_bounds__(256) void hg_wcvt_t(const float* __restrict__ W,
                                                 unsigned short* __restrict__ Wt,
                                                 int K, int N) {
  size_t i = (size_t)blockIdx.x * 256 + threadIdx.x;
  if (i >= (size_t)K * N) return;
  int k = (int)(i / N), n = (int)(i % N);
  Wt[(size_t)n * K + k] = f2bf(W[i]);
}

// ---------------------------------------------------------------------------
// Elementwise / norm kernels
// ---------------------------------------------------------------------------
__global__ __launch_bounds__(256) void hg_copy(const float* __restrict__ a,
                                               float* __restrict__ b, size_t n) {
  size_t i = (size_t)blockIdx.x * 256 + threadIdx.x;
  if (i < n) b[i] = a[i];
}

__global__ __launch_bounds__(256) void hg_mix(const float* __restrict__ x,
                                              const float* __restrict__ x0,
                                              const float* __restrict__ m, // [2][D]
                                              float* __restrict__ xin,
                                              int D, size_t n) {
  size_t i = (size_t)blockIdx.x * 256 + threadIdx.x;
  if (i >= n) return;
  int col = (int)(i % D);
  xin[i] = m[col] * x[i] + m[D + col] * x0[i];
}

__global__ __launch_bounds__(256) void hg_addscale(const float* __restrict__ base,
                                                   const float* __restrict__ scale,
                                                   const float* __restrict__ delta,
                                                   float* __restrict__ out,
                                                   int D, size_t n) {
  size_t i = (size_t)blockIdx.x * 256 + threadIdx.x;
  if (i >= n) return;
  int col = (int)(i % D);
  out[i] = base[i] + scale[col] * delta[i];
}

// relu^2 with bf16 output (feeds the down-projection GEMM)
__global__ __launch_bounds__(256) void hg_relu2cvt(const float* __restrict__ x,
                                                   unsigned short* __restrict__ o,
                                                   size_t n) {
  size_t i = (size_t)blockIdx.x * 256 + threadIdx.x;
  if (i >= n) return;
  float v = fmaxf(x[i], 0.f);
  o[i] = f2bf(v * v);
}

__global__ __launch_bounds__(256) void hg_sigmoid(float* __restrict__ x, size_t n) {
  size_t i = (size_t)blockIdx.x * 256 + threadIdx.x;
  if (i >= n) return;
  x[i] = 1.f / (1.f + __expf(-x[i]));
}

// row-wise RMS norm over D, bf16 output (feeds projection GEMMs)
__global__ __launch_bounds__(256) void hg_rmsrow(const float* __restrict__ x,
                                                 unsigned short* __restrict__ y,
                                                 int D) {
  const int row = blockIdx.x;
  const float* xr = x + (size_t)row * D;
  unsigned short* yr = y + (size_t)row * D;
  float ss = 0.f;
  for (int i = threadIdx.x; i < D; i += 256) { float v = xr[i]; ss += v * v; }
  #pragma unroll
  for (int off = 16; off > 0; off >>= 1) ss += __shfl_xor(ss, off, 32);
  __shared__ float red[8];
  if ((threadIdx.x & 31) == 0) red[threadIdx.x >> 5] = ss;
  __syncthreads();
  float tot = 0.f;
  #pragma unroll
  for (int i = 0; i < 8; ++i) tot += red[i];
  const float rs = rsqrtf(tot / (float)D + 1e-6f);
  for (int i = threadIdx.x; i < D; i += 256) yr[i] = f2bf(xr[i] * rs);
}

// per-head (128-wide) norm, in place fp32. useMean=1 -> rms, 0 -> l2n
__global__ __launch_bounds__(128) void hg_headnorm(float* __restrict__ x, int useMean) {
  const size_t base = (size_t)blockIdx.x * 128;
  const int j = threadIdx.x;
  float v = x[base + j];
  float ss = v * v;
  #pragma unroll
  for (int off = 16; off > 0; off >>= 1) ss += __shfl_xor(ss, off, 32);
  __shared__ float red[4];
  if ((j & 31) == 0) red[j >> 5] = ss;
  __syncthreads();
  float tot = red[0] + red[1] + red[2] + red[3];
  float d = useMean ? (tot * (1.f / 128.f)) : tot;
  x[base + j] = v * rsqrtf(d + 1e-6f);
}

// rope (+ optional per-head gain), in place on [.., heads, 128]
__global__ __launch_bounds__(64) void hg_rope(float* __restrict__ x,
                                              const float* __restrict__ gain,
                                              int T, int heads) {
  const int hh = blockIdx.x % heads;
  const int bt = blockIdx.x / heads;
  const int t = bt % T;
  float* p = x + ((size_t)bt * heads + hh) * 128;
  const int i = threadIdx.x;                 // 0..63 (pair index)
  const float inv = __expf(-(2.f * (float)i / 128.f) * 9.210340371976184f);
  const float fr = (float)t * inv;
  const float c = __cosf(fr), s = __sinf(fr);
  const float g = gain ? gain[hh] : 1.f;
  const float x1 = p[i], x2 = p[i + 64];
  p[i]      = (x1 * c - x2 * s) * g;
  p[i + 64] = (x2 * c + x1 * s) * g;
}

// gated-deltanet output gate: out = bf16( rms_head(o) * silu(g) )
__global__ __launch_bounds__(128) void hg_gdn_gate(const float* __restrict__ o,
                                                   const float* __restrict__ g,
                                                   unsigned short* __restrict__ out) {
  const size_t base = (size_t)blockIdx.x * 128;
  const int j = threadIdx.x;
  float ov = o[base + j];
  float ss = ov * ov;
  #pragma unroll
  for (int off = 16; off > 0; off >>= 1) ss += __shfl_xor(ss, off, 32);
  __shared__ float red[4];
  if ((j & 31) == 0) red[j >> 5] = ss;
  __syncthreads();
  float tot = red[0] + red[1] + red[2] + red[3];
  float rn = ov * rsqrtf(tot * (1.f / 128.f) + 1e-6f);
  float gv = g[base + j];
  out[base + j] = f2bf(rn * (gv / (1.f + __expf(-gv))));
}

// ---------------------------------------------------------------------------
// GDN delta-rule scan. One block per (b,h); thread j owns state column S[:,j]
// (128 floats register-resident). k/q broadcast to LDS via async DMA.
// ---------------------------------------------------------------------------
__global__ __launch_bounds__(128) void hg_gdn_scan(
    const float* __restrict__ q, const float* __restrict__ k,
    const float* __restrict__ v, const float* __restrict__ a,
    const float* __restrict__ be, float* __restrict__ o,
    int T, int H)
{
  const int h = blockIdx.x % H;
  const int b = blockIdx.x / H;
  const int j = threadIdx.x;
  float s[128];
  #pragma unroll
  for (int i = 0; i < 128; ++i) s[i] = 0.f;
  __shared__ float sh[256];                 // [0..127]=k, [128..255]=q
  const unsigned lk = (unsigned)(j * 4);
  const unsigned lq = (unsigned)(512 + j * 4);
  for (int t = 0; t < T; ++t) {
    const size_t base = (((size_t)b * T + t) * H + h) * 128;
    {
      const float* gk = k + base + j;
      const float* gq = q + base + j;
      asm volatile("global_load_async_to_lds_b32 %0, %1, off"
                   :: "v"(lk), "v"(gk) : "memory");
      asm volatile("global_load_async_to_lds_b32 %0, %1, off"
                   :: "v"(lq), "v"(gq) : "memory");
      asm volatile("s_wait_asynccnt 0x0" ::: "memory");
    }
    __syncthreads();
    const float at = a[((size_t)b * T + t) * H + h];
    const float bt = be[((size_t)b * T + t) * H + h];
    const float vt = v[base + j];
    float kS = 0.f;
    #pragma unroll
    for (int i = 0; i < 128; ++i) { s[i] *= at; kS += sh[i] * s[i]; }
    const float dl = bt * (vt - kS);
    float ov = 0.f;
    #pragma unroll
    for (int i = 0; i < 128; ++i) { s[i] += sh[i] * dl; ov += sh[128 + i] * s[i]; }
    o[base + j] = ov;
    __syncthreads();
  }
}

// ---------------------------------------------------------------------------
// Sliding-window flash attention with WMMA. One wave per 16-query tile.
// q:[B,T,H,128] f32 (rms+rope+gain), k,v:[B,T,HKV,128] f32, y bf16 [B,T,H,128]
// ---------------------------------------------------------------------------
__global__ __launch_bounds__(32) void hg_swa_attn(
    const float* __restrict__ q, const float* __restrict__ k,
    const float* __restrict__ v, unsigned short* __restrict__ y,
    int T, int H, int HKV)
{
  const int ntile = T / 16;
  const int qt  = blockIdx.x % ntile;
  const int rem = blockIdx.x / ntile;
  const int h = rem % H;
  const int b = rem / H;
  const int hkv = h / (H / HKV);
  const int lane = threadIdx.x;
  const int l15 = lane & 15;
  const int half = (lane >> 4) & 1;
  const int tq0 = qt * 16;

  __shared__ float          Sf[16][32];
  __shared__ unsigned short Pb[16][32];
  __shared__ unsigned short Vt[128][32 + 2];
  __shared__ float          rowtmp[16];

  bf16x16 qf[4];
  {
    const size_t qbase = (((size_t)b * T + (tq0 + l15)) * H + h) * 128;
    #pragma unroll
    for (int c = 0; c < 4; ++c) {
      const int d0 = c * 32 + half * 16;
      u16x16 tv;
      #pragma unroll
      for (int i = 0; i < 8; ++i) {
        tv[i]     = f2bf(q[qbase + d0 + i]);
        tv[8 + i] = f2bf(q[qbase + d0 + 8 + i]);
      }
      qf[c] = __builtin_bit_cast(bf16x16, tv);
    }
  }

  f32x8 oacc[8];
  #pragma unroll
  for (int i = 0; i < 8; ++i) oacc[i] = (f32x8){};
  float mrow = -1e30f, lrow = 0.f;

  int kstart = tq0 - 511; if (kstart < 0) kstart = 0;
  kstart &= ~31;
  const float scale = 0.08838834764831845f; // 1/sqrt(128)

  for (int ks = kstart; ks <= tq0 + 15; ks += 32) {
    f32x8 sacc0 = {}; f32x8 sacc1 = {};
    #pragma unroll
    for (int c = 0; c < 4; ++c) {
      const int d0 = c * 32 + half * 16;
      int key0 = ks + l15;       if (key0 >= T) key0 = T - 1;
      int key1 = ks + 16 + l15;  if (key1 >= T) key1 = T - 1;
      const size_t kb0 = (((size_t)b * T + key0) * HKV + hkv) * 128 + d0;
      const size_t kb1 = (((size_t)b * T + key1) * HKV + hkv) * 128 + d0;
      u16x16 b0v, b1v;
      #pragma unroll
      for (int i = 0; i < 16; ++i) { b0v[i] = f2bf(k[kb0 + i]); b1v[i] = f2bf(k[kb1 + i]); }
      bf16x16 b0f = __builtin_bit_cast(bf16x16, b0v);
      bf16x16 b1f = __builtin_bit_cast(bf16x16, b1v);
      sacc0 = __builtin_amdgcn_wmma_f32_16x16x32_bf16(false, qf[c], false, b0f, (short)0, sacc0, false, false);
      sacc1 = __builtin_amdgcn_wmma_f32_16x16x32_bf16(false, qf[c], false, b1f, (short)0, sacc1, false, false);
    }
    #pragma unroll
    for (int i = 0; i < 8; ++i) {
      const int row = half * 8 + i;
      Sf[row][l15]      = sacc0[i] * scale;
      Sf[row][16 + l15] = sacc1[i] * scale;
    }
    __syncthreads();

    {
      const int row = l15;
      const int tqr = tq0 + row;
      float pmax = -1e30f;
      float sv[16];
      #pragma unroll
      for (int i = 0; i < 16; ++i) {
        const int col = half * 16 + i;
        const int key = ks + col;
        const bool ok = (key >= 0) && (key <= tqr) && ((tqr - key) < 512);
        float s0 = ok ? Sf[row][col] : -1e30f;
        sv[i] = s0;
        pmax = fmaxf(pmax, s0);
      }
      pmax = fmaxf(pmax, __shfl_xor(pmax, 16, 32));
      const float mnew = fmaxf(mrow, pmax);
      const float alpha = __expf(mrow - mnew);
      float psum = 0.f;
      #pragma unroll
      for (int i = 0; i < 16; ++i) {
        float p = __expf(sv[i] - mnew);
        psum += p;
        Pb[row][half * 16 + i] = f2bf(p);
      }
      psum += __shfl_xor(psum, 16, 32);
      lrow = lrow * alpha + psum;
      mrow = mnew;
      if (half == 0) rowtmp[row] = alpha;
    }
    __syncthreads();

    {
      float al[8];
      #pragma unroll
      for (int i = 0; i < 8; ++i) al[i] = rowtmp[half * 8 + i];
      #pragma unroll
      for (int nt = 0; nt < 8; ++nt)
        #pragma unroll
        for (int i = 0; i < 8; ++i) oacc[nt][i] *= al[i];
    }

    {
      const int key = ks + lane;
      int keyc = key; if (keyc < 0) keyc = 0; if (keyc >= T) keyc = T - 1;
      const bool ok = (key >= 0) && (key < T);
      const size_t vb0 = (((size_t)b * T + keyc) * HKV + hkv) * 128;
      for (int d = 0; d < 128; ++d) {
        float vv = ok ? v[vb0 + d] : 0.f;
        Vt[d][lane] = f2bf(vv);
      }
    }
    __syncthreads();

    {
      const int k0 = half * 16;
      u16x16 av;
      #pragma unroll
      for (int i = 0; i < 8; ++i) { av[i] = Pb[l15][k0 + i]; av[8 + i] = Pb[l15][k0 + 8 + i]; }
      bf16x16 af = __builtin_bit_cast(bf16x16, av);
      #pragma unroll
      for (int nt = 0; nt < 8; ++nt) {
        const unsigned short* vp = &Vt[nt * 16 + l15][half * 16];
        u16x16 bv;
        #pragma unroll
        for (int i = 0; i < 16; ++i) bv[i] = vp[i];
        bf16x16 bfv = __builtin_bit_cast(bf16x16, bv);
        oacc[nt] = __builtin_amdgcn_wmma_f32_16x16x32_bf16(false, af, false, bfv, (short)0, oacc[nt], false, false);
      }
    }
    __syncthreads();
  }

  if (half == 0) rowtmp[l15] = 1.f / lrow;
  __syncthreads();
  {
    float li[8];
    #pragma unroll
    for (int i = 0; i < 8; ++i) li[i] = rowtmp[half * 8 + i];
    #pragma unroll
    for (int nt = 0; nt < 8; ++nt) {
      #pragma unroll
      for (int i = 0; i < 8; ++i) {
        const int row = tq0 + half * 8 + i;
        y[(((size_t)b * T + row) * H + h) * 128 + nt * 16 + l15] = f2bf(oacc[nt][i] * li[i]);
      }
    }
  }
}

// ---------------------------------------------------------------------------
// Host orchestration
// ---------------------------------------------------------------------------
static inline unsigned grd(size_t n) { return (unsigned)((n + 255) / 256); }

extern "C" void kernel_launch(void* const* d_in, const int* in_sizes, int n_in,
                              void* d_out, int out_size, void* d_ws, size_t ws_size,
                              hipStream_t stream) {
  const int Bc = 2, Tc = 2048, Dc = 1024, Hc = 8, DHc = 128, HKVc = 4;
  const size_t NT = (size_t)Bc * Tc;      // 4096 rows
  const size_t ND = NT * Dc;              // 4M elements

  const float* x0    = (const float*)d_in[0];
  const float* gWq   = (const float*)d_in[1];
  const float* gWk   = (const float*)d_in[2];
  const float* gWv   = (const float*)d_in[3];
  const float* gWa   = (const float*)d_in[4];
  const float* gWb   = (const float*)d_in[5];
  const float* gWg   = (const float*)d_in[6];
  const float* gWo   = (const float*)d_in[7];
  const float* gfc   = (const float*)d_in[8];
  const float* gpj   = (const float*)d_in[9];
  const float* gasc  = (const float*)d_in[10];
  const float* gmsc  = (const float*)d_in[11];
  const float* gmix  = (const float*)d_in[12];
  const float* sWq   = (const float*)d_in[13];
  const float* sWk   = (const float*)d_in[14];
  const float* sWv   = (const float*)d_in[15];
  const float* sWo   = (const float*)d_in[16];
  const float* sgain = (const float*)d_in[17];
  const float* sfc   = (const float*)d_in[18];
  const float* spj   = (const float*)d_in[19];
  const float* sasc  = (const float*)d_in[20];
  const float* smsc  = (const float*)d_in[21];
  const float* smix  = (const float*)d_in[22];

  // ---- workspace layout: fp32 region then bf16 region ----
  float* f = (float*)d_ws;
  float* xb   = f; f += ND;     // residual stream
  float* xin  = f; f += ND;     // mixed input / xo (in place)
  float* qb   = f; f += ND;
  float* kb   = f; f += ND;
  float* vb   = f; f += ND;
  float* gb   = f; f += ND;
  float* ob   = f; f += ND;     // scan output
  float* mo   = f; f += ND;     // mixer / mlp GEMM output
  float* hid  = f; f += 3 * ND; // fc output (fp32)
  float* aa   = f; f += NT * Hc;
  float* bbuf = f; f += NT * Hc;
  unsigned short* u = (unsigned short*)f;
  unsigned short* hbb  = u; u += ND;      // rms output (bf16 GEMM A)
  unsigned short* hidb = u; u += 3 * ND;  // relu^2 output (bf16 GEMM A)
  unsigned short* gob  = u; u += ND;      // gate / attention output (bf16 GEMM A)
  unsigned short* wp   = u;               // bf16 transposed weights

  auto cvtw = [&](const float* W, int K, int N) -> const unsigned short* {
    unsigned short* dst = wp;
    size_t elems = (size_t)K * N;
    hg_wcvt_t<<<grd(elems), 256, 0, stream>>>(W, dst, K, N);
    wp += elems;
    return dst;
  };

  // pre-convert + transpose all weights to bf16 (deterministic layout)
  const unsigned short *cWq[4], *cWk[4], *cWv[4], *cWg[4], *cWo[4];
  const unsigned short *cWa[4], *cWb[4], *cfc[4], *cpj[4];
  for (int i = 0; i < 4; ++i) {
    cWq[i] = cvtw(gWq + (size_t)i * Dc * Dc, Dc, Dc);
    cWk[i] = cvtw(gWk + (size_t)i * Dc * Dc, Dc, Dc);
    cWv[i] = cvtw(gWv + (size_t)i * Dc * Dc, Dc, Dc);
    cWg[i] = cvtw(gWg + (size_t)i * Dc * Dc, Dc, Dc);
    cWo[i] = cvtw(gWo + (size_t)i * Dc * Dc, Dc, Dc);
    cWa[i] = cvtw(gWa + (size_t)i * Dc * Hc, Dc, Hc);
    cWb[i] = cvtw(gWb + (size_t)i * Dc * Hc, Dc, Hc);
    cfc[i] = cvtw(gfc + (size_t)i * Dc * 3 * Dc, Dc, 3 * Dc);
    cpj[i] = cvtw(gpj + (size_t)i * 3 * Dc * Dc, 3 * Dc, Dc);
  }
  const unsigned short *sWqc[2], *sWkc[2], *sWvc[2], *sWoc[2], *sfcc[2], *spjc[2];
  for (int i = 0; i < 2; ++i) {
    sWqc[i] = cvtw(sWq + (size_t)i * Dc * Dc, Dc, Dc);
    sWkc[i] = cvtw(sWk + (size_t)i * Dc * (HKVc * DHc), Dc, HKVc * DHc);
    sWvc[i] = cvtw(sWv + (size_t)i * Dc * (HKVc * DHc), Dc, HKVc * DHc);
    sWoc[i] = cvtw(sWo + (size_t)i * Dc * Dc, Dc, Dc);
    sfcc[i] = cvtw(sfc + (size_t)i * Dc * 3 * Dc, Dc, 3 * Dc);
    spjc[i] = cvtw(spj + (size_t)i * 3 * Dc * Dc, 3 * Dc, Dc);
  }

  auto gemm = [&](const unsigned short* A, const unsigned short* Wt, float* C,
                  int M, int N, int K) {
    dim3 g((N + TN - 1) / TN, (M + TM - 1) / TM);
    hg_gemm<<<g, 256, 0, stream>>>(A, Wt, C, M, N, K);
  };

  hg_copy<<<grd(ND), 256, 0, stream>>>(x0, xb, ND);

  const char layout[6] = {'g', 'g', 's', 'g', 'g', 's'};
  int gi = 0, si = 0;
  for (int L = 0; L < 6; ++L) {
    if (layout[L] == 'g') {
      const int i = gi++;
      hg_mix<<<grd(ND), 256, 0, stream>>>(xb, x0, gmix + (size_t)i * 2 * Dc, xin, Dc, ND);
      hg_rmsrow<<<(unsigned)NT, 256, 0, stream>>>(xin, hbb, Dc);
      gemm(hbb, cWq[i], qb, (int)NT, Dc, Dc);
      gemm(hbb, cWk[i], kb, (int)NT, Dc, Dc);
      gemm(hbb, cWv[i], vb, (int)NT, Dc, Dc);
      gemm(hbb, cWg[i], gb, (int)NT, Dc, Dc);
      gemm(hbb, cWa[i], aa, (int)NT, Hc, Dc);
      gemm(hbb, cWb[i], bbuf, (int)NT, Hc, Dc);
      hg_sigmoid<<<grd(NT * Hc), 256, 0, stream>>>(aa, NT * Hc);
      hg_sigmoid<<<grd(NT * Hc), 256, 0, stream>>>(bbuf, NT * Hc);
      hg_headnorm<<<(unsigned)(NT * Hc), 128, 0, stream>>>(qb, 0);  // l2n
      hg_headnorm<<<(unsigned)(NT * Hc), 128, 0, stream>>>(kb, 0);  // l2n
      hg_gdn_scan<<<Bc * Hc, 128, 0, stream>>>(qb, kb, vb, aa, bbuf, ob, Tc, Hc);
      hg_gdn_gate<<<(unsigned)(NT * Hc), 128, 0, stream>>>(ob, gb, gob);
      gemm(gob, cWo[i], mo, (int)NT, Dc, Dc);
      hg_addscale<<<grd(ND), 256, 0, stream>>>(xin, gasc + (size_t)i * Dc, mo, xin, Dc, ND);
      // MLP
      hg_rmsrow<<<(unsigned)NT, 256, 0, stream>>>(xin, hbb, Dc);
      gemm(hbb, cfc[i], hid, (int)NT, 3 * Dc, Dc);
      hg_relu2cvt<<<grd(NT * 3 * Dc), 256, 0, stream>>>(hid, hidb, NT * 3 * Dc);
      gemm(hidb, cpj[i], mo, (int)NT, Dc, 3 * Dc);
      hg_addscale<<<grd(ND), 256, 0, stream>>>(xin, gmsc + (size_t)i * Dc, mo, xb, Dc, ND);
    } else {
      const int i = si++;
      hg_mix<<<grd(ND), 256, 0, stream>>>(xb, x0, smix + (size_t)i * 2 * Dc, xin, Dc, ND);
      hg_rmsrow<<<(unsigned)NT, 256, 0, stream>>>(xin, hbb, Dc);
      gemm(hbb, sWqc[i], qb, (int)NT, Dc, Dc);
      gemm(hbb, sWkc[i], kb, (int)NT, HKVc * DHc, Dc);
      gemm(hbb, sWvc[i], vb, (int)NT, HKVc * DHc, Dc);
      hg_headnorm<<<(unsigned)(NT * Hc), 128, 0, stream>>>(qb, 1);   // rms
      hg_headnorm<<<(unsigned)(NT * HKVc), 128, 0, stream>>>(kb, 1); // rms
      hg_rope<<<(unsigned)(NT * Hc), 64, 0, stream>>>(qb, sgain + (size_t)i * Hc, Tc, Hc);
      hg_rope<<<(unsigned)(NT * HKVc), 64, 0, stream>>>(kb, nullptr, Tc, HKVc);
      hg_swa_attn<<<(unsigned)(Bc * Hc * (Tc / 16)), 32, 0, stream>>>(qb, kb, vb, gob, Tc, Hc, HKVc);
      gemm(gob, sWoc[i], mo, (int)NT, Dc, Dc);
      hg_addscale<<<grd(ND), 256, 0, stream>>>(xin, sasc + (size_t)i * Dc, mo, xin, Dc, ND);
      // MLP
      hg_rmsrow<<<(unsigned)NT, 256, 0, stream>>>(xin, hbb, Dc);
      gemm(hbb, sfcc[i], hid, (int)NT, 3 * Dc, Dc);
      hg_relu2cvt<<<grd(NT * 3 * Dc), 256, 0, stream>>>(hid, hidb, NT * 3 * Dc);
      gemm(hidb, spjc[i], mo, (int)NT, Dc, 3 * Dc);
      hg_addscale<<<grd(ND), 256, 0, stream>>>(xin, smsc + (size_t)i * Dc, mo, xb, Dc, ND);
    }
  }
  hg_copy<<<grd(ND), 256, 0, stream>>>(xb, (float*)d_out, ND);
}
